// FactorizedTrilinear_16192026706010
// MI455X (gfx1250) — matrix operand order repaired and checked by reading, not compiled, not run on hardware
//
#include <hip/hip_runtime.h>

typedef __attribute__((ext_vector_type(16))) _Float16 v16h;
typedef __attribute__((ext_vector_type(8)))  float    v8f;

#define NB 4
#define NZ 128
#define NIN 512
#define NR 256

// workspace layout: 3 tensors of [16 inst (b*4+d*2+v)][128 rows][256 e] f16
#define INST_STRIDE (128 * 256)
#define TENSOR_STRIDE (16 * INST_STRIDE)

// ---------------- projection: p = x @ W + b, stored f16, re-laid-out --------
__global__ __launch_bounds__(256) void ft_proj_kernel(
    const float* __restrict__ x1, const float* __restrict__ x2, const float* __restrict__ x3,
    const float* __restrict__ W1, const float* __restrict__ b1,
    const float* __restrict__ W2, const float* __restrict__ b2,
    const float* __restrict__ W3, const float* __restrict__ b3,
    _Float16* __restrict__ P)
{
    __shared__ float xs[NIN];
    const int t = blockIdx.x / 2048;   // which tensor
    const int r = blockIdx.x % 2048;   // flat (b,z,d,v) row
    const float* xsrc = (t == 0 ? x1 : (t == 1 ? x2 : x3)) + (size_t)r * NIN;
    const float* W    = (t == 0 ? W1 : (t == 1 ? W2 : W3));
    const float* bias = (t == 0 ? b1 : (t == 1 ? b2 : b3));

    const int tid = threadIdx.x;
    xs[tid]       = xsrc[tid];
    xs[tid + 256] = xsrc[tid + 256];
    __syncthreads();

    // r = ((b*Z + z)*D + d)*V + v
    const int v = r & 1, d = (r >> 1) & 1, z = (r >> 2) & 127, b = r >> 9;
    const int inst = (b * 2 + d) * 2 + v;

    float acc = bias[tid];
#pragma unroll 8
    for (int i = 0; i < NIN; ++i)
        acc = fmaf(xs[i], W[i * NR + tid], acc);

    P[(size_t)t * TENSOR_STRIDE + (size_t)inst * INST_STRIDE + (size_t)z * NR + tid] =
        (_Float16)acc;
}

// ---------------- trilinear contraction via WMMA ----------------------------
union FragH { v16h h; uint4 q[2]; };

__global__ __launch_bounds__(256) void ft_tri_kernel(
    const _Float16* __restrict__ P, float* __restrict__ out)
{
    // blockIdx.x = ((b*8 + zt)*8 + xt)*8 + ct
    const int blk = blockIdx.x;
    const int ct = blk & 7, xt = (blk >> 3) & 7, zt = (blk >> 6) & 7, b = blk >> 9;
    const int z0 = zt * 16, x0 = xt * 16, c0 = ct * 16;

    const int lane = threadIdx.x & 31;
    const int wave = threadIdx.x >> 5;
    const int hl = lane & 15;      // M (A) or N (B/C/D)
    const int hi = lane >> 4;      // lane half selects K sub-range / M+8

    const _Float16* P1 = P;
    const _Float16* P2 = P + TENSOR_STRIDE;
    const _Float16* P3 = P + 2 * (size_t)TENSOR_STRIDE;

    v8f acc[2][4];
#pragma unroll
    for (int cc = 0; cc < 2; ++cc)
#pragma unroll
        for (int dv = 0; dv < 4; ++dv)
            acc[cc][dv] = (v8f){0.f, 0.f, 0.f, 0.f, 0.f, 0.f, 0.f, 0.f};

#pragma unroll
    for (int dv = 0; dv < 4; ++dv) {
        const size_t inst = (size_t)(b * 4 + dv) * INST_STRIDE;
        const _Float16* p1row = P1 + inst + (size_t)(z0 + hl) * NR;  // A: row M = hl
        const _Float16* p2row = P2 + inst + (size_t)(x0 + hl) * NR;  // B: col N = hl
        const _Float16* p3ins = P3 + inst;

        // Hoist A and p2 fragments for the whole K=256 sweep (8 chunks of 32)
        FragH a[8], p2f[8];
#pragma unroll
        for (int kc = 0; kc < 8; ++kc) {
            // A k order used here: lane half hi -> K 0..15 or 16..31 (contiguous, matches B)
            const _Float16* ap = p1row + kc * 32 + hi * 16;
            a[kc].q[0] = *(const uint4*)(ap);        // K +0..7
            a[kc].q[1] = *(const uint4*)(ap + 8);    // K +8..15
            // B k order used here: lane half hi -> K 0..15 or 16..31 (contiguous, matches A)
            const _Float16* bp = p2row + kc * 32 + hi * 16;
            p2f[kc].q[0] = *(const uint4*)(bp);
            p2f[kc].q[1] = *(const uint4*)(bp + 8);
        }

#pragma unroll
        for (int cc = 0; cc < 2; ++cc) {
            const int c = c0 + wave * 2 + cc;
            const _Float16* p3row = p3ins + (size_t)c * NR;
#pragma unroll
            for (int kc = 0; kc < 8; ++kc) {
                FragH p3f, bm;
                const _Float16* cp = p3row + kc * 32 + hi * 16;
                p3f.q[0] = *(const uint4*)(cp);
                p3f.q[1] = *(const uint4*)(cp + 8);
                bm.h = p2f[kc].h * p3f.h;            // v_pk_mul_f16: B = p2 ⊙ p3[c]
                acc[cc][dv] = __builtin_amdgcn_wmma_f32_16x16x32_f16(
                    false, a[kc].h, false, bm.h, (short)0, acc[cc][dv], false, false);
            }
        }
    }

    // out[b][z][x][c][d][v]: (c,d,v) contiguous -> pack 4 dv planes into b128 stores
#pragma unroll
    for (int cc = 0; cc < 2; ++cc) {
        const int c = c0 + wave * 2 + cc;
        const size_t base = (size_t)b * 8388608 + (size_t)(x0 + hl) * 512 + (size_t)c * 4;
#pragma unroll
        for (int i = 0; i < 8; ++i) {
            const int zz = z0 + i + hi * 8;          // C/D layout: M = i + 8*hi
            float4 val = make_float4(acc[cc][0][i], acc[cc][1][i],
                                     acc[cc][2][i], acc[cc][3][i]);
            *(float4*)(out + base + (size_t)zz * 65536) = val;
        }
    }
}

extern "C" void kernel_launch(void* const* d_in, const int* in_sizes, int n_in,
                              void* d_out, int out_size, void* d_ws, size_t ws_size,
                              hipStream_t stream) {
    const float* x1 = (const float*)d_in[0];
    const float* x2 = (const float*)d_in[1];
    const float* x3 = (const float*)d_in[2];
    const float* W1 = (const float*)d_in[3];
    const float* b1 = (const float*)d_in[4];
    const float* W2 = (const float*)d_in[5];
    const float* b2 = (const float*)d_in[6];
    const float* W3 = (const float*)d_in[7];
    const float* b3 = (const float*)d_in[8];

    _Float16* P = (_Float16*)d_ws;   // 3 MB of f16 projections
    float* out = (float*)d_out;

    ft_proj_kernel<<<3 * 2048, 256, 0, stream>>>(x1, x2, x3, W1, b1, W2, b2, W3, b3, P);
    ft_tri_kernel<<<2048, 256, 0, stream>>>(P, out);
}
